// Block_21869973471768
// MI455X (gfx1250) — matrix operand order repaired and checked by reading, not compiled, not run on hardware
//
#include <hip/hip_runtime.h>
#include <hip/hip_bf16.h>

// ---------------------------------------------------------------------------
// Transformer block (B=4, N=1024, D=1024, H=16, Dh=64) for gfx1250 (CDNA5).
// All GEMMs run on v_wmma_f32_16x16x32_bf16 (f32 accumulate).
// ---------------------------------------------------------------------------

typedef __bf16 bf16;
typedef __attribute__((ext_vector_type(16))) __bf16 v16bf;
typedef __attribute__((ext_vector_type(8)))  __bf16 v8bf;
typedef __attribute__((ext_vector_type(8)))  float  v8f;

#define DEVINL __device__ __forceinline__

DEVINL v16bf join8(v8bf lo, v8bf hi) {
    union { v16bf v; v8bf h[2]; } u;
    u.h[0] = lo; u.h[1] = hi;
    return u.v;
}

DEVINL v8f wmma_bf16(v16bf a, v16bf b, v8f c) {
    return __builtin_amdgcn_wmma_f32_16x16x32_bf16(
        /*neg_a=*/false, a, /*neg_b=*/false, b,
        /*c_mod=*/(short)0, c, /*reuse_a=*/false, /*reuse_b=*/false);
}

// ---------------------------------------------------------------------------
// Weight convert + transpose via LDS tiles (coalesced both ways):
// W fp32 [K,N] row-major -> Wt bf16 [N,K] row-major.
// Block (32,8), one 32x32 tile per block.
// ---------------------------------------------------------------------------
__global__ __launch_bounds__(256)
void wtrans_kernel(const float* __restrict__ W, bf16* __restrict__ Wt,
                   int K, int N) {
    __shared__ bf16 tile[32][33];
    const int k0 = blockIdx.x * 32;
    const int n0 = blockIdx.y * 32;
    const int tx = threadIdx.x;   // 0..31
    const int ty = threadIdx.y;   // 0..7
#pragma unroll
    for (int j = 0; j < 4; j++) {
        int k = k0 + ty + j * 8;
        tile[ty + j * 8][tx] = (bf16)W[(size_t)k * N + n0 + tx];
    }
    __syncthreads();
#pragma unroll
    for (int j = 0; j < 4; j++) {
        int n = n0 + ty + j * 8;
        Wt[(size_t)n * K + k0 + tx] = tile[tx][ty + j * 8];
    }
}

// ---------------------------------------------------------------------------
// LayerNorm over rows of length 1024, fp32 in -> bf16 out. One block per row.
// ---------------------------------------------------------------------------
__global__ __launch_bounds__(128)
void ln_kernel(const float* __restrict__ x, const float* __restrict__ w,
               const float* __restrict__ bvec, bf16* __restrict__ out) {
    const int row = blockIdx.x;
    const int tid = threadIdx.x;
    const float* xr = x + (size_t)row * 1024;

    float vals[8];
    float s = 0.f, s2 = 0.f;
#pragma unroll
    for (int t = 0; t < 8; t++) {
        float v = xr[t * 128 + tid];
        vals[t] = v; s += v; s2 += v * v;
    }
#pragma unroll
    for (int off = 16; off > 0; off >>= 1) {
        s  += __shfl_xor(s,  off, 32);
        s2 += __shfl_xor(s2, off, 32);
    }
    __shared__ float ps[4], ps2[4];
    int wv = tid >> 5;
    if ((tid & 31) == 0) { ps[wv] = s; ps2[wv] = s2; }
    __syncthreads();
    s  = ps[0]  + ps[1]  + ps[2]  + ps[3];
    s2 = ps2[0] + ps2[1] + ps2[2] + ps2[3];

    float mu  = s * (1.0f / 1024.0f);
    float var = s2 * (1.0f / 1024.0f) - mu * mu;
    float rs  = rsqrtf(var + 1e-5f);

    bf16* orow = out + (size_t)row * 1024;
#pragma unroll
    for (int t = 0; t < 8; t++) {
        int i = t * 128 + tid;
        orow[i] = (bf16)((vals[t] - mu) * rs * w[i] + bvec[i]);
    }
}

// ---------------------------------------------------------------------------
// WMMA GEMM: C[M,N] = A[M,K](bf16) x Wt^T (Wt is [N,K] bf16) + bias, fused
// epilogues. 32x64 tile per wave (8 wmma per K-step of 32), 4 waves per
// block covering 128 rows x 64 cols.
// ---------------------------------------------------------------------------
enum GemmMode { MODE_QK = 0, MODE_VT = 1, MODE_RES = 2, MODE_GELU = 3 };

template <int MODE>
__global__ __launch_bounds__(128)
void gemm_kernel(const bf16* __restrict__ A, const bf16* __restrict__ Wt,
                 const float* __restrict__ bias, void* __restrict__ outp,
                 const float* __restrict__ resid, int M, int N, int K) {
    const int lane = threadIdx.x & 31;
    const int wave = threadIdx.x >> 5;
    const int l15  = lane & 15;
    const int half = lane >> 4;
    const int rowBase = blockIdx.y * 128 + wave * 32;
    const int colBase = blockIdx.x * 64;

    v8f acc[2][4];
#pragma unroll
    for (int i = 0; i < 2; i++)
#pragma unroll
        for (int j = 0; j < 4; j++)
#pragma unroll
            for (int r = 0; r < 8; r++) acc[i][j][r] = 0.0f;

    for (int kb = 0; kb < K; kb += 32) {
        v16bf af[2], bfr[4];
#pragma unroll
        for (int ms = 0; ms < 2; ms++) {
            // A layout: lane=row (l15), elems j -> K=half*16+j
            const bf16* ap = A + (size_t)(rowBase + ms * 16 + l15) * K + kb + half * 16;
            af[ms] = join8(*(const v8bf*)ap, *(const v8bf*)(ap + 8));
        }
#pragma unroll
        for (int ns = 0; ns < 4; ns++) {
            // B layout: lane=column (l15), elems j -> K = half*16 + j (contiguous in Wt row)
            const bf16* bp = Wt + (size_t)(colBase + ns * 16 + l15) * K + kb + half * 16;
            bfr[ns] = *(const v16bf*)bp;
        }
#pragma unroll
        for (int ms = 0; ms < 2; ms++)
#pragma unroll
            for (int ns = 0; ns < 4; ns++)
                acc[ms][ns] = wmma_bf16(af[ms], bfr[ns], acc[ms][ns]);
    }

    // Epilogue. C layout: element r of acc -> (row = r + 8*half, col = l15) per subtile
#pragma unroll
    for (int ms = 0; ms < 2; ms++) {
#pragma unroll
        for (int ns = 0; ns < 4; ns++) {
#pragma unroll
            for (int r = 0; r < 8; r++) {
                int R = rowBase + ms * 16 + half * 8 + r;
                int C = colBase + ns * 16 + l15;
                float v = acc[ms][ns][r] + bias[C];
                if constexpr (MODE == MODE_QK) {
                    // out[b, h, tok, d] bf16
                    int b = R >> 10, tok = R & 1023;
                    int hh = C >> 6, d = C & 63;
                    ((bf16*)outp)[(((size_t)(b * 16 + hh)) * 1024 + tok) * 64 + d] = (bf16)v;
                } else if constexpr (MODE == MODE_VT) {
                    // out[b, h, d, tok] bf16 (transposed V for WMMA B-operand)
                    int b = R >> 10, tok = R & 1023;
                    int hh = C >> 6, d = C & 63;
                    ((bf16*)outp)[(((size_t)(b * 16 + hh)) * 64 + d) * 1024 + tok] = (bf16)v;
                } else if constexpr (MODE == MODE_RES) {
                    size_t idx = (size_t)R * N + C;
                    ((float*)outp)[idx] = v + resid[idx];
                } else { // MODE_GELU: exact GELU, bf16 out
                    float g = 0.5f * v * (1.0f + erff(v * 0.70710678118654752f));
                    ((bf16*)outp)[(size_t)R * N + C] = (bf16)g;
                }
            }
        }
    }
}

// ---------------------------------------------------------------------------
// Flash attention: one wave per (b, h, 16-query tile). Non-causal softmax.
// QK^T and PV both on WMMA; P re-striped C-layout -> A-layout through LDS.
// ---------------------------------------------------------------------------
__global__ __launch_bounds__(32)
void attn_kernel(const bf16* __restrict__ q, const bf16* __restrict__ k,
                 const bf16* __restrict__ vt, bf16* __restrict__ o) {
    const int lane = threadIdx.x;
    const int l15  = lane & 15;
    const int half = lane >> 4;
    const int qt = blockIdx.x;   // 0..63 (16 queries each)
    const int h  = blockIdx.y;   // head
    const int b  = blockIdx.z;   // batch
    const int bh = b * 16 + h;
    const bf16* qp = q  + (size_t)bh * 1024 * 64;
    const bf16* kp = k  + (size_t)bh * 1024 * 64;
    const bf16* vp = vt + (size_t)bh * 64 * 1024;
    const int q0 = qt * 16;

    __shared__ __align__(16) bf16 sp[16 * 40]; // 16 q-rows x 32 keys, stride 40

    // Preload Q A-fragments (Dh=64 -> two K-steps of 32)
    v16bf qa[2];
#pragma unroll
    for (int ks = 0; ks < 2; ks++) {
        const bf16* ap = qp + (size_t)(q0 + l15) * 64 + ks * 32 + half * 16;
        qa[ks] = join8(*(const v8bf*)ap, *(const v8bf*)(ap + 8));
    }

    float mrun[8], lrun[8];
    v8f oacc[4];
#pragma unroll
    for (int r = 0; r < 8; r++) { mrun[r] = -3.0e38f; lrun[r] = 0.0f; }
#pragma unroll
    for (int d4 = 0; d4 < 4; d4++)
#pragma unroll
        for (int r = 0; r < 8; r++) oacc[d4][r] = 0.0f;

    const float SC = 0.125f * 1.44269504088896340736f; // scale * log2(e)

    for (int kb = 0; kb < 32; kb++) { // 32 key-blocks of 32 keys
        v8f sc[2];
#pragma unroll
        for (int ns = 0; ns < 2; ns++)
#pragma unroll
            for (int r = 0; r < 8; r++) sc[ns][r] = 0.0f;

#pragma unroll
        for (int ks = 0; ks < 2; ks++) {
#pragma unroll
            for (int ns = 0; ns < 2; ns++) {
                // B-fragment of K^T: lane = key column, elems j -> d = ks*32+half*16+j
                const bf16* bp = kp + (size_t)(kb * 32 + ns * 16 + l15) * 64 + ks * 32 + half * 16;
                v16bf kf = *(const v16bf*)bp;
                sc[ns] = wmma_bf16(qa[ks], kf, sc[ns]);
            }
        }

        // Online softmax per owned row (row m = half*8 + r, cols striped on 16 lanes)
#pragma unroll
        for (int r = 0; r < 8; r++) {
            float t = fmaxf(sc[0][r], sc[1][r]);
#pragma unroll
            for (int off = 8; off > 0; off >>= 1) t = fmaxf(t, __shfl_xor(t, off, 32));
            float newm  = fmaxf(mrun[r], t);
            float alpha = exp2f((mrun[r] - newm) * SC);
            mrun[r] = newm;
            float p0 = exp2f((sc[0][r] - newm) * SC);
            float p1 = exp2f((sc[1][r] - newm) * SC);
            float rs = p0 + p1;
#pragma unroll
            for (int off = 8; off > 0; off >>= 1) rs += __shfl_xor(rs, off, 32);
            lrun[r] = lrun[r] * alpha + rs;
#pragma unroll
            for (int d4 = 0; d4 < 4; d4++) oacc[d4][r] *= alpha;
            int m = half * 8 + r;
            sp[m * 40 + l15]      = (bf16)p0;
            sp[m * 40 + 16 + l15] = (bf16)p1;
        }
        __syncthreads();

        // Re-read P as A-fragment (lane=row l15, K=key: half*16+j)
        v16bf pa;
        {
            const bf16* pp = sp + l15 * 40 + half * 16;
            pa = join8(*(const v8bf*)pp, *(const v8bf*)(pp + 8));
        }
#pragma unroll
        for (int d4 = 0; d4 < 4; d4++) {
            // B-fragment of V: lane = d column, elems j -> key = kb*32+half*16+j
            const bf16* bp = vp + (size_t)(d4 * 16 + l15) * 1024 + kb * 32 + half * 16;
            v16bf vf = *(const v16bf*)bp;
            oacc[d4] = wmma_bf16(pa, vf, oacc[d4]);
        }
        __syncthreads();
    }

    // Normalize and scatter to o[b, tok, h*64+d] bf16 (row-major [4096,1024])
#pragma unroll
    for (int r = 0; r < 8; r++) {
        float inv = 1.0f / lrun[r];
        int tok = q0 + half * 8 + r;
        bf16* orow = o + ((size_t)(b * 1024 + tok)) * 1024 + h * 64;
#pragma unroll
        for (int d4 = 0; d4 < 4; d4++)
            orow[d4 * 16 + l15] = (bf16)(oacc[d4][r] * inv);
    }
}

// ---------------------------------------------------------------------------
// Launch
// ---------------------------------------------------------------------------
extern "C" void kernel_launch(void* const* d_in, const int* in_sizes, int n_in,
                              void* d_out, int out_size, void* d_ws, size_t ws_size,
                              hipStream_t stream) {
    (void)in_sizes; (void)n_in; (void)out_size; (void)ws_size;
    const float* x     = (const float*)d_in[0];
    const float* ln1w  = (const float*)d_in[1];
    const float* ln1b  = (const float*)d_in[2];
    const float* Wq    = (const float*)d_in[3];
    const float* bq    = (const float*)d_in[4];
    const float* Wk    = (const float*)d_in[5];
    const float* bk    = (const float*)d_in[6];
    const float* Wv    = (const float*)d_in[7];
    const float* bv    = (const float*)d_in[8];
    const float* Wo    = (const float*)d_in[9];
    const float* bo    = (const float*)d_in[10];
    const float* ln2w  = (const float*)d_in[11];
    const float* ln2b  = (const float*)d_in[12];
    const float* Wfc   = (const float*)d_in[13];
    const float* bfc   = (const float*)d_in[14];
    const float* Wproj = (const float*)d_in[15];
    const float* bproj = (const float*)d_in[16];

    char* ws = (char*)d_ws;
    const size_t MB = 1024 * 1024;
    bf16*  wqt  = (bf16*)(ws + 0 * MB);   // 2 MB each
    bf16*  wkt  = (bf16*)(ws + 2 * MB);
    bf16*  wvt  = (bf16*)(ws + 4 * MB);
    bf16*  wot  = (bf16*)(ws + 6 * MB);
    bf16*  wfct = (bf16*)(ws + 8 * MB);   // 8 MB  [4096,1024]
    bf16*  wprt = (bf16*)(ws + 16 * MB);  // 8 MB  [1024,4096]
    bf16*  h1   = (bf16*)(ws + 24 * MB);  // 8 MB
    bf16*  qb   = (bf16*)(ws + 32 * MB);  // 8 MB
    bf16*  kbuf = (bf16*)(ws + 40 * MB);  // 8 MB
    bf16*  vtb  = (bf16*)(ws + 48 * MB);  // 8 MB
    bf16*  ob   = (bf16*)(ws + 56 * MB);  // 8 MB
    float* x1   = (float*)(ws + 64 * MB); // 16 MB
    bf16*  h2   = (bf16*)(ws + 24 * MB);  // alias h1 (dead)
    bf16*  fcb  = (bf16*)(ws + 32 * MB);  // 32 MB, alias q/k/vt/o (dead)

    // Convert + transpose weights to bf16 (LDS-tiled, coalesced)
    dim3 tb(32, 8);
    wtrans_kernel<<<dim3(1024 / 32, 1024 / 32), tb, 0, stream>>>(Wq, wqt, 1024, 1024);
    wtrans_kernel<<<dim3(1024 / 32, 1024 / 32), tb, 0, stream>>>(Wk, wkt, 1024, 1024);
    wtrans_kernel<<<dim3(1024 / 32, 1024 / 32), tb, 0, stream>>>(Wv, wvt, 1024, 1024);
    wtrans_kernel<<<dim3(1024 / 32, 1024 / 32), tb, 0, stream>>>(Wo, wot, 1024, 1024);
    wtrans_kernel<<<dim3(1024 / 32, 4096 / 32), tb, 0, stream>>>(Wfc, wfct, 1024, 4096);
    wtrans_kernel<<<dim3(4096 / 32, 1024 / 32), tb, 0, stream>>>(Wproj, wprt, 4096, 1024);

    // LN1
    ln_kernel<<<4096, 128, 0, stream>>>(x, ln1w, ln1b, h1);

    // QKV projections (blocks cover 128 rows x 64 cols)
    dim3 g1(1024 / 64, 4096 / 128);
    gemm_kernel<MODE_QK><<<g1, 128, 0, stream>>>(h1, wqt, bq, qb,   nullptr, 4096, 1024, 1024);
    gemm_kernel<MODE_QK><<<g1, 128, 0, stream>>>(h1, wkt, bk, kbuf, nullptr, 4096, 1024, 1024);
    gemm_kernel<MODE_VT><<<g1, 128, 0, stream>>>(h1, wvt, bv, vtb,  nullptr, 4096, 1024, 1024);

    // Attention
    dim3 ga(64, 16, 4);
    attn_kernel<<<ga, 32, 0, stream>>>(qb, kbuf, vtb, ob);

    // Output projection + residual (fp32)
    gemm_kernel<MODE_RES><<<g1, 128, 0, stream>>>(ob, wot, bo, (void*)x1, x, 4096, 1024, 1024);

    // LN2
    ln_kernel<<<4096, 128, 0, stream>>>(x1, ln2w, ln2b, h2);

    // FC + exact GELU (bf16 out)
    dim3 g2(4096 / 64, 4096 / 128);
    gemm_kernel<MODE_GELU><<<g2, 128, 0, stream>>>(h2, wfct, bfc, fcb, nullptr, 4096, 4096, 1024);

    // Proj + residual -> final fp32 output
    gemm_kernel<MODE_RES><<<g1, 128, 0, stream>>>(fcb, wprt, bproj, d_out, x1, 4096, 1024, 4096);
}
